// MultiDimHiPPO2_89172111000297
// MI455X (gfx1250) — matrix operand order repaired and checked by reading, not compiled, not run on hardware
//
#include <hip/hip_runtime.h>

// ---------------------------------------------------------------------------
// MultiDimHiPPO trig-basis GEMM for gfx1250 (MI455X).
//   out[b,r,m] = base[b,r,m] + sum_k trig(2*pi*(k+tok)*(r&511)/8192) * slice[b,k,m]
// Batched GEMM (1024 x 4096) x (4096 x 1024), batch 4, via v_wmma_f32_16x16x32_f16.
// A built from a half-period 4096-entry f16 LDS trig table (sign via phase bit).
// B staged global->LDS with gfx1250 async copies (ASYNCcnt), double-buffered,
// one barrier per K-chunk.
// ---------------------------------------------------------------------------

typedef _Float16 v16h __attribute__((ext_vector_type(16)));
typedef float    v8f  __attribute__((ext_vector_type(8)));

#define TWO_PI_F 6.283185307179586f

#define B_BATCH 4
#define K_TOT   4096
#define M_TOT   1024
#define NROWS   1024      // 512 cos rows + 512 sin rows

#define KC      32        // K per WMMA step (f16 16x16x32)
#define NT      128       // output rows per block
#define MT      128       // output cols per block

#define LDA_H   40        // A tile: 32 halves/row padded to 40 (80 B, 16B-aligned)
#define LDBF    132       // B tile: 128 floats/row padded to 132 (528 B, 16B-aligned)

__global__ __launch_bounds__(256)
void hippo_trig_gemm_wmma(const float* __restrict__ base,
                          const float* __restrict__ slice,
                          const int*   __restrict__ token_ptr,
                          float*       __restrict__ out)
{
    // LDS: 8192 + 2*10240 + 2*16896 = 62464 B
    __shared__ __align__(16) unsigned short trig_h[4096];       // half-period table
    __shared__ __align__(16) _Float16 lds_a[2][NT * LDA_H];     // f16 A tiles
    __shared__ __align__(16) float    lds_b[2][KC * LDBF];      // f32 B tiles (async dst)

    const int t   = threadIdx.x;
    const int bx  = blockIdx.x;          // m tile   (0..7)
    const int by  = blockIdx.y;          // row tile (0..7): 0-3 cos, 4-7 sin
    const int bz  = blockIdx.z;          // batch    (0..3)
    const int tok = token_ptr[0];

    // ---- half-period trig table: trig(2*pi*i/8192), i in [0,4096) ----------
    const bool isSin = (by >= 4);
    for (int i = t; i < 4096; i += 256) {
        float ang = (float)i * (TWO_PI_F / 8192.0f);
        float v   = isSin ? __sinf(ang) : __cosf(ang);
        _Float16 h = (_Float16)v;
        union { _Float16 f; unsigned short u; } cv; cv.f = h;
        trig_h[i] = cv.u;
    }

    const int wid  = t >> 5;             // wave 0..7 (wave32)
    const int lane = t & 31;
    const int lhi  = lane >> 4;
    const int llo  = lane & 15;

    // staging assignments
    const int arow   = t >> 1;           // 0..127 : A tile row
    const int ahalf  = (t & 1) * 16;     // 0/16   : k sub-chunk
    const int n_freq = (by * NT + arow) & 511;
    const int brow   = t >> 3;           // 0..31  : B tile k-row
    const int bcol   = (t & 7) * 16;     // 0..112 : B tile col base (floats)

    // async B staging: SGPR 64-bit base + per-lane 32-bit byte offset (GVS)
    const float* gB = slice + (size_t)bz * K_TOT * M_TOT;                  // uniform
    const unsigned gofs0 = (unsigned)(((brow * M_TOT) + bx * MT + bcol) * 4);
    const unsigned lofsB[2] = {
        (unsigned)(size_t)(&lds_b[0][brow * LDBF + bcol]),   // LDS addr = flat[31:0]
        (unsigned)(size_t)(&lds_b[1][brow * LDBF + bcol])
    };

    auto stage_b_async = [&](int kc, int nxt) {
        const unsigned g = gofs0 + (unsigned)(kc * (M_TOT * 4));
        const unsigned l = lofsB[nxt];
        asm volatile("global_load_async_to_lds_b128 %0, %1, %2 offset:0"
                     :: "v"(l), "v"(g), "s"(gB) : "memory");
        asm volatile("global_load_async_to_lds_b128 %0, %1, %2 offset:16"
                     :: "v"(l), "v"(g), "s"(gB) : "memory");
        asm volatile("global_load_async_to_lds_b128 %0, %1, %2 offset:32"
                     :: "v"(l), "v"(g), "s"(gB) : "memory");
        asm volatile("global_load_async_to_lds_b128 %0, %1, %2 offset:48"
                     :: "v"(l), "v"(g), "s"(gB) : "memory");
    };

    auto stage_a = [&](int kc, int nxt) {
        const int kmod = (kc + ahalf + tok) & 8191;
        int phase = (kmod * n_freq) & 8191;
        union { unsigned short u[16]; uint4 q[2]; } au;
        #pragma unroll
        for (int j = 0; j < 16; ++j) {
            unsigned short v = trig_h[phase & 4095];
            v = (unsigned short)(v ^ ((phase & 4096) << 3));  // +pi -> sign flip
            au.u[j] = v;
            phase = (phase + n_freq) & 8191;
        }
        uint4* dst = (uint4*)(&lds_a[nxt][arow * LDA_H + ahalf]);
        dst[0] = au.q[0];
        dst[1] = au.q[1];
    };

    // wave sub-tile 64x32: 4 (row) x 2 (col) f32 accumulators
    v8f acc[4][2];
    #pragma unroll
    for (int r = 0; r < 4; ++r)
        #pragma unroll
        for (int c = 0; c < 2; ++c)
            #pragma unroll
            for (int i = 0; i < 8; ++i)
                acc[r][c][i] = 0.0f;

    __syncthreads();                 // trig table ready
    stage_b_async(0, 0);
    stage_a(0, 0);
    asm volatile("s_wait_asynccnt 0x0" ::: "memory");
    __syncthreads();                 // buffers[0] ready

    auto body = [&](int kc, int cur) {
        const int nxt = cur ^ 1;
        if (kc + KC < K_TOT) {       // prefetch next chunk into other buffer
            stage_b_async(kc + KC, nxt);
            stage_a(kc + KC, nxt);
        }
        // ---- A fragments (ISA 7.12.2 16-bit A 16x32 layout) ----
        v16h afr[4];
        const _Float16* aB = lds_a[cur];
        #pragma unroll
        for (int r = 0; r < 4; ++r) {
            const int row = (wid >> 2) * 64 + r * 16 + llo;
            const uint4* p = (const uint4*)(aB + row * LDA_H);
            union { uint4 q[2]; v16h v; } u;
            u.q[0] = p[lhi];
            u.q[1] = p[2 + lhi];
            afr[r] = u.v;
        }
        // ---- B fragments: f32 LDS -> f16 (N = lane%16; K table order) ----
        v16h bfr[2];
        const float* bB = lds_b[cur] + llo;
        #pragma unroll
        for (int c = 0; c < 2; ++c) {
            const int n0 = (wid & 3) * 32 + c * 16;
            const float* pc = bB + n0;
            union { _Float16 h[16]; v16h v; } u;
            #pragma unroll
            for (int s = 0; s < 16; ++s)
                u.h[s] = (_Float16)pc[(s < 8 ? 8 * lhi + s : 16 + 8 * lhi + (s - 8)) * LDBF];
            bfr[c] = u.v;
        }
        // ---- 8 x v_wmma_f32_16x16x32_f16 ----
        #pragma unroll
        for (int r = 0; r < 4; ++r)
            #pragma unroll
            for (int c = 0; c < 2; ++c)
                acc[r][c] = __builtin_amdgcn_wmma_f32_16x16x32_f16(
                    false, afr[r], false, bfr[c],
                    (short)0, acc[r][c], false, false);

        asm volatile("s_wait_asynccnt 0x0" ::: "memory");  // next B tile landed
        __syncthreads();                                   // next A stores + cur reads done
    };

    for (int kc = 0; kc < K_TOT; kc += 2 * KC) {   // unroll x2: buffer sel compile-time
        body(kc, 0);
        body(kc + KC, 1);
    }

    // ---- epilogue: D layout (VGPR i: M = i + 8*lhi, N = llo) + residual ----
    #pragma unroll
    for (int r = 0; r < 4; ++r) {
        #pragma unroll
        for (int c = 0; c < 2; ++c) {
            const int col = bx * MT + (wid & 3) * 32 + c * 16 + llo;
            #pragma unroll
            for (int i = 0; i < 8; ++i) {
                const int row = by * NT + (wid >> 2) * 64 + r * 16 + lhi * 8 + i;
                const size_t o = ((size_t)bz * NROWS + row) * M_TOT + col;
                out[o] = base[o] + acc[r][c][i];
            }
        }
    }
}

extern "C" void kernel_launch(void* const* d_in, const int* in_sizes, int n_in,
                              void* d_out, int out_size, void* d_ws, size_t ws_size,
                              hipStream_t stream) {
    (void)in_sizes; (void)n_in; (void)out_size; (void)d_ws; (void)ws_size;
    const float* base  = (const float*)d_in[0];   // (4, 1024, 1024) f32
    const float* slice = (const float*)d_in[1];   // (4, 4096, 1024) f32
    const int*   tokp  = (const int*)d_in[2];     // scalar token_num
    float*       out   = (float*)d_out;           // (4, 1024, 1024) f32

    dim3 grid(M_TOT / MT, NROWS / NT, B_BATCH);   // 8 x 8 x 4 = 256 blocks
    dim3 block(256);                              // 8 wave32 per block
    hippo_trig_gemm_wmma<<<grid, block, 0, stream>>>(base, slice, tokp, out);
}
